// ws_Net_9698036154472
// MI455X (gfx1250) — compile-verified, matrix operand order checked by reading, not run on hardware
//
#include <hip/hip_runtime.h>
#include <hip/hip_bf16.h>

typedef __attribute__((ext_vector_type(8)))  _Float16 v8h;
typedef __attribute__((ext_vector_type(16))) _Float16 v16h;
typedef __attribute__((ext_vector_type(8)))  float    v8f;

// ---------------------------------------------------------------------------
// Problem constants
// ---------------------------------------------------------------------------
#define BATCH   128
#define NT      390            // timesteps
#define ROWS    (BATCH * NT)   // 49920 fused (b,t) rows
#define K1      1008           // fc1 K
#define K1P     1024           // fc1 K padded to multiple of 32
#define N1      50             // fc1 N
#define K2      19500          // fc2 K  (= 390*50)
#define K2P     19520          // fc2 K padded (610*32)
#define N2      512            // fc2 N

// workspace layout (bytes)
#define WS_FC1H   0                            // [64][1024] f16      131072 B
#define WS_FC2H   (WS_FC1H + 64*1024*2)        // [512][19520] f16  19988480 B
#define WS_Y1     (WS_FC2H + 512*19520*2)      // [128][19520] f16   4997120 B
#define WS_Y2     (WS_Y1   + 128*19520*2)      // [128][512]  f32     262144 B

// ---------------------------------------------------------------------------
// WMMA fragment loader: row-major matrix, 16 rows x (K-chunk of 32) halfs.
// Per CDNA5 ISA layout: lane l<16 holds row l, K = kbase+{0..7, 16..23};
// lane 16+l holds row l, K = kbase+{8..15, 24..31}.  Two 16B vector loads.
// ---------------------------------------------------------------------------
__device__ __forceinline__ v16h load_frag16(const _Float16* base, int ld, int kbase) {
    const int lane = threadIdx.x & 31;
    const int row  = lane & 15;
    const int kh   = (lane >> 4) << 3;  // 0 or 8
    const _Float16* p = base + row * ld + kbase + kh;
    v8h lo = *(const v8h*)(p);
    v8h hi = *(const v8h*)(p + 16);
    v16h out;
#pragma unroll
    for (int i = 0; i < 8; ++i) { out[i] = lo[i]; out[i + 8] = hi[i]; }
    return out;
}

__device__ __forceinline__ float relu(float v) { return v > 0.f ? v : 0.f; }

// Async global->LDS copy (CDNA5): LDS dest address is low 32 bits of the
// generic shared pointer; tracked by ASYNCcnt, no VGPR round-trip.
__device__ __forceinline__ void async_load_b32(const void* gaddr, void* ldsaddr) {
    unsigned           lo = (unsigned)(unsigned long long)(uintptr_t)ldsaddr;
    unsigned long long ga = (unsigned long long)(uintptr_t)gaddr;
    asm volatile("global_load_async_to_lds_b32 %0, %1, off"
                 :: "v"(lo), "v"(ga) : "memory");
}
__device__ __forceinline__ void wait_async0() {
    asm volatile("s_wait_asynccnt 0x0" ::: "memory");
}

// ---------------------------------------------------------------------------
// Kernel 0: weight prep (f32 -> f16, zero-padded) + zero fc2-input K-pad cols
// ---------------------------------------------------------------------------
__global__ void __launch_bounds__(256)
prep_kernel(const float* __restrict__ fc1w, const float* __restrict__ fc2w,
            _Float16* __restrict__ fc1h, _Float16* __restrict__ fc2h,
            _Float16* __restrict__ y1) {
    const int i = blockIdx.x * 256 + threadIdx.x;
    const int A = 64 * K1P;            // 65536
    const int B = 512 * K2P;           // 9994240
    const int C = 128 * (K2P - K2);    // 2560
    if (i < A) {
        int n = i >> 10, k = i & 1023;
        float v = (n < N1 && k < K1) ? fc1w[n * K1 + k] : 0.f;
        fc1h[i] = (_Float16)v;
    } else if (i < A + B) {
        int j = i - A;
        int n = j / K2P, k = j - n * K2P;
        float v = (k < K2) ? fc2w[n * K2 + k] : 0.f;
        fc2h[j] = (_Float16)v;
    } else if (i < A + B + C) {
        int j = i - (A + B);
        int b = j / 20, k = j - b * 20;
        y1[b * K2P + K2 + k] = (_Float16)0.f;
    }
}

// ---------------------------------------------------------------------------
// Kernel 1: fused conv1+conv2+conv3 (VALU, fp32) + fc1 (WMMA f16->f32)
// One workgroup (4 waves) handles 16 (b,t) rows.
// ---------------------------------------------------------------------------
__global__ void __launch_bounds__(128)
conv_fc1_kernel(const float* __restrict__ x,
                const float* __restrict__ c1w, const float* __restrict__ c1b,
                const float* __restrict__ c2w, const float* __restrict__ c2b,
                const float* __restrict__ c3w, const float* __restrict__ c3b,
                const _Float16* __restrict__ fc1h, const float* __restrict__ fc1b,
                _Float16* __restrict__ y1) {
    // static LDS: 2304 + 16512 + 32768 = 51584 B
    __shared__ __align__(16) float    wS[576];          // conv weights/biases
    __shared__ __align__(16) float    xS[8 * 516];      // 8 rows of x slice [4][129]
    __shared__ __align__(16) _Float16 aS[16 * K1P];     // fc1 A-tile, 16 x 1024

    const int tid  = threadIdx.x;
    const int wave = tid >> 5;
    const int lane = tid & 31;
    const int grB  = blockIdx.x * 16;   // first global row of this tile
    const int b0   = grB / NT;          // uniform scalar division
    const int t0   = grB - b0 * NT;     // 16-row tile wraps batch at most once

    // --- load conv weights into LDS (broadcast-friendly) ---
    // [0..31] c1w  [32..39] c1b  [40..295] c2w  [296..303] c2b
    // [304..559] c3w  [560..567] c3b
    for (int i = tid; i < 568; i += 128) {
        float v;
        if      (i <  32) v = c1w[i];
        else if (i <  40) v = c1b[i - 32];
        else if (i < 296) v = c2w[i - 40];
        else if (i < 304) v = c2b[i - 296];
        else if (i < 560) v = c3w[i - 304];
        else              v = c3b[i - 560];
        wS[i] = v;
    }
    // --- zero K-pad columns 1008..1023 of the A-tile ---
    for (int i = tid; i < 16 * 16; i += 128)
        aS[(i >> 4) * K1P + K1 + (i & 15)] = (_Float16)0.f;

    // --- two half-batches of 8 rows: stage x async, run convs, fill A-tile ---
    for (int half = 0; half < 2; ++half) {
        __syncthreads();
        // x element (b, h, w, t) sits at ((b*4+h)*129 + w)*390 + t ;
        // consecutive m => consecutive t => consecutive global addresses.
        for (int i = tid; i < 8 * 516; i += 128) {
            int m = i & 7;
            int e = i >> 3;                  // 0..515
            int h = e / 129, w = e - h * 129;
            int t = t0 + half * 8 + m;
            int b = b0;
            if (t >= NT) { t -= NT; b += 1; }
            async_load_b32(x + ((b * 4 + h) * 129 + w) * NT + t,
                           xS + m * 516 + e);
        }
        wait_async0();
        __syncthreads();

        // each wave computes 2 rows; each lane sweeps output positions w3
        for (int rr = 0; rr < 2; ++rr) {
            const int mloc = wave * 2 + rr;          // 0..7
            const int m    = half * 8 + mloc;        // 0..15 tile row
            const float* X = xS + mloc * 516;
            for (int p = 0; p < 4; ++p) {
                const int w3 = lane + 32 * p;        // conv3 output column
                if (w3 >= 126) break;

                float xv[4][4];
#pragma unroll
                for (int h = 0; h < 4; ++h)
#pragma unroll
                    for (int j = 0; j < 4; ++j)
                        xv[h][j] = X[h * 129 + w3 + j];

                // conv1: a1[c1][h1][dw], h1 in 0..2, dw in 0..2
                float a1[8][3][3];
#pragma unroll
                for (int c1 = 0; c1 < 8; ++c1) {
                    const float k00 = wS[c1 * 4 + 0], k01 = wS[c1 * 4 + 1];
                    const float k10 = wS[c1 * 4 + 2], k11 = wS[c1 * 4 + 3];
                    const float bb  = wS[32 + c1];
#pragma unroll
                    for (int h1 = 0; h1 < 3; ++h1)
#pragma unroll
                        for (int dw = 0; dw < 3; ++dw)
                            a1[c1][h1][dw] = relu(bb
                                + k00 * xv[h1][dw]     + k01 * xv[h1][dw + 1]
                                + k10 * xv[h1 + 1][dw] + k11 * xv[h1 + 1][dw + 1]);
                }

                // conv2: a2[c2][h2][dw2], h2,dw2 in 0..1
                float a2[8][2][2];
#pragma unroll
                for (int c2 = 0; c2 < 8; ++c2) {
                    const float bb = wS[296 + c2];
                    float s00 = bb, s01 = bb, s10 = bb, s11 = bb;
#pragma unroll
                    for (int c1 = 0; c1 < 8; ++c1) {
                        const float* wp = wS + 40 + c2 * 32 + c1 * 4;
                        const float k00 = wp[0], k01 = wp[1], k10 = wp[2], k11 = wp[3];
                        s00 += k00*a1[c1][0][0] + k01*a1[c1][0][1] + k10*a1[c1][1][0] + k11*a1[c1][1][1];
                        s01 += k00*a1[c1][0][1] + k01*a1[c1][0][2] + k10*a1[c1][1][1] + k11*a1[c1][1][2];
                        s10 += k00*a1[c1][1][0] + k01*a1[c1][1][1] + k10*a1[c1][2][0] + k11*a1[c1][2][1];
                        s11 += k00*a1[c1][1][1] + k01*a1[c1][1][2] + k10*a1[c1][2][1] + k11*a1[c1][2][2];
                    }
                    a2[c2][0][0] = relu(s00); a2[c2][0][1] = relu(s01);
                    a2[c2][1][0] = relu(s10); a2[c2][1][1] = relu(s11);
                }

                // conv3 -> flatten index c3*126 + w3, f16 into A-tile
#pragma unroll
                for (int c3 = 0; c3 < 8; ++c3) {
                    float acc = wS[560 + c3];
#pragma unroll
                    for (int c2 = 0; c2 < 8; ++c2) {
                        const float* wp = wS + 304 + c3 * 32 + c2 * 4;
                        acc += wp[0]*a2[c2][0][0] + wp[1]*a2[c2][0][1]
                             + wp[2]*a2[c2][1][0] + wp[3]*a2[c2][1][1];
                    }
                    aS[m * K1P + c3 * 126 + w3] = (_Float16)relu(acc);
                }
            }
        }
    }
    __syncthreads();

    // --- fc1 via WMMA: each wave owns one 16-wide N-tile, K = 1024 ---
    const _Float16* Bp = fc1h + wave * 16 * K1P;
    v8f acc = {};
#pragma unroll 4
    for (int k = 0; k < K1P; k += 32) {
        v16h af = load_frag16(aS, K1P, k);
        v16h bf = load_frag16(Bp, K1P, k);
        acc = __builtin_amdgcn_wmma_f32_16x16x32_f16(
            false, af, false, bf, (short)0, acc, false, false);
    }

    // epilogue: D row = r + 8*(lane>=16), col = lane&15 ; bias + ReLU.
    // (b,t) from uniform (b0,t0) + at most one wrap -- no per-lane division.
    const int ncol   = wave * 16 + (lane & 15);
    const int rowadd = (lane >> 4) * 8;
    if (ncol < N1) {
        const float bias = fc1b[ncol];
#pragma unroll
        for (int r = 0; r < 8; ++r) {
            int t = t0 + r + rowadd;
            int b = b0;
            if (t >= NT) { t -= NT; b += 1; }
            y1[b * K2P + t * N1 + ncol] = (_Float16)relu(acc[r] + bias);
        }
    }
}

// ---------------------------------------------------------------------------
// Kernel 2: fc2 GEMM [128 x 19520] x [19520 x 512] via WMMA, bias + ReLU.
// One 16x16 tile per single-wave workgroup: 256 WGs spread across WGPs.
// ---------------------------------------------------------------------------
__global__ void __launch_bounds__(32)
fc2_kernel(const _Float16* __restrict__ y1, const _Float16* __restrict__ fc2h,
           const float* __restrict__ fc2b, float* __restrict__ y2) {
    const int lane = threadIdx.x & 31;
    const int mt   = blockIdx.x >> 5;     // 0..7
    const int nt   = blockIdx.x & 31;     // 0..31

    const _Float16* Ap = y1   + mt * 16 * K2P;
    const _Float16* Bp = fc2h + nt * 16 * K2P;

    v8f acc = {};
#pragma unroll 2
    for (int k = 0; k < K2P; k += 32) {
        // keep the L2 stream ahead of the matrix pipe
        __builtin_prefetch(Ap + ((lane & 15) * K2P) + k + 256, 0, 1);
        v16h af = load_frag16(Ap, K2P, k);
        v16h bf = load_frag16(Bp, K2P, k);
        acc = __builtin_amdgcn_wmma_f32_16x16x32_f16(
            false, af, false, bf, (short)0, acc, false, false);
    }

    const int n      = nt * 16 + (lane & 15);
    const int rowadd = (lane >> 4) * 8;
    const float bias = fc2b[n];
#pragma unroll
    for (int r = 0; r < 8; ++r) {
        const int m = mt * 16 + r + rowadd;
        y2[m * N2 + n] = relu(acc[r] + bias);
    }
}

// ---------------------------------------------------------------------------
// Kernel 3: fc3 [128 x 512] x [512 x 2] + softmax (tiny; 1 thread per row)
// ---------------------------------------------------------------------------
__global__ void __launch_bounds__(128)
fc3_softmax_kernel(const float* __restrict__ y2, const float* __restrict__ fc3w,
                   const float* __restrict__ fc3b, float* __restrict__ out) {
    const int b = blockIdx.x * blockDim.x + threadIdx.x;
    if (b >= BATCH) return;
    const float* r = y2 + b * N2;
    float a0 = fc3b[0], a1 = fc3b[1];
    for (int k = 0; k < N2; ++k) {
        const float v = r[k];
        a0 += v * fc3w[k];
        a1 += v * fc3w[N2 + k];
    }
    const float mx = fmaxf(a0, a1);
    const float e0 = expf(a0 - mx), e1 = expf(a1 - mx);
    const float inv = 1.f / (e0 + e1);
    out[b * 2 + 0] = e0 * inv;
    out[b * 2 + 1] = e1 * inv;
}

// ---------------------------------------------------------------------------
extern "C" void kernel_launch(void* const* d_in, const int* in_sizes, int n_in,
                              void* d_out, int out_size, void* d_ws, size_t ws_size,
                              hipStream_t stream) {
    const float* x    = (const float*)d_in[0];
    // d_in[1] = n (int, 390) — hardcoded
    const float* c1w  = (const float*)d_in[2];
    const float* c1b  = (const float*)d_in[3];
    const float* c2w  = (const float*)d_in[4];
    const float* c2b  = (const float*)d_in[5];
    const float* c3w  = (const float*)d_in[6];
    const float* c3b  = (const float*)d_in[7];
    const float* fc1w = (const float*)d_in[8];
    const float* fc1b = (const float*)d_in[9];
    const float* fc2w = (const float*)d_in[10];
    const float* fc2b = (const float*)d_in[11];
    const float* fc3w = (const float*)d_in[12];
    const float* fc3b = (const float*)d_in[13];
    float* out = (float*)d_out;

    char* ws = (char*)d_ws;
    _Float16* fc1h = (_Float16*)(ws + WS_FC1H);
    _Float16* fc2h = (_Float16*)(ws + WS_FC2H);
    _Float16* y1   = (_Float16*)(ws + WS_Y1);
    float*    y2   = (float*)   (ws + WS_Y2);

    // 65536 + 9994240 + 2560 = 10062336 elements -> 39306 blocks x 256
    prep_kernel<<<39306, 256, 0, stream>>>(fc1w, fc2w, fc1h, fc2h, y1);

    conv_fc1_kernel<<<ROWS / 16, 128, 0, stream>>>(
        x, c1w, c1b, c2w, c2b, c3w, c3b, fc1h, fc1b, y1);

    fc2_kernel<<<256, 32, 0, stream>>>(y1, fc2h, fc2b, y2);

    fc3_softmax_kernel<<<1, 128, 0, stream>>>(y2, fc3w, fc3b, out);
}
